// PCGen_38104949850213
// MI455X (gfx1250) — matrix operand order repaired and checked by reading, not compiled, not run on hardware
//
#include <hip/hip_runtime.h>
#include <hip/hip_bf16.h>

typedef __attribute__((ext_vector_type(16))) __bf16 v16bf;
typedef __attribute__((ext_vector_type(8)))  __bf16 v8bf;
typedef __attribute__((ext_vector_type(8)))  float  v8f;

#define N_TOTAL 65536   // B*M
#define MPTS    2048
#define NB      32

// LDS byte offset of a __shared__ object (flat LDS addresses carry the offset in the low 32 bits)
#define LDS_OFF(p) ((unsigned)(size_t)(void*)(p))

// ---------------- workspace layout (bytes) ----------------
static constexpr size_t SZ_BUF   = (size_t)N_TOTAL * 512 * 2;     // 64 MB bf16 activation buffer
static constexpr size_t OFF_BUFA = 0;
static constexpr size_t OFF_BUFB = SZ_BUF;
static constexpr size_t OFF_WM1  = 2 * SZ_BUF;                    // 256x32 bf16 (padded)
static constexpr size_t OFF_WM2  = OFF_WM1 + 256 * 32 * 2;        // 512x256
static constexpr size_t OFF_WC0  = OFF_WM2 + 512 * 256 * 2;       // 512x512
static constexpr size_t OFF_WC1  = OFF_WC0 + 512 * 512 * 2;       // 256x512
static constexpr size_t OFF_WC2  = OFF_WC1 + 256 * 512 * 2;       // 128x256
static constexpr size_t OFF_WC3  = OFF_WC2 + 128 * 256 * 2;       // 64x128
static constexpr size_t OFF_STAT = OFF_WC3 + 64 * 128 * 2;        // float[4][2][512] sum/sumsq
static constexpr size_t OFF_BN   = OFF_STAT + 4 * 2 * 512 * 4;    // float[4][2][512] scale/shift
static constexpr size_t OFF_PTS  = OFF_BN + 4 * 2 * 512 * 4;      // float[N_TOTAL*3]

// ---------------- small utility kernels ----------------
__global__ void zero_stats_kernel(float* p, int n) {
    int i = blockIdx.x * blockDim.x + threadIdx.x;
    if (i < n) p[i] = 0.0f;
}

__global__ void cvt_kernel(const float* __restrict__ src, __bf16* __restrict__ dst, int n) {
    int i = blockIdx.x * blockDim.x + threadIdx.x;
    if (i < n) dst[i] = (__bf16)src[i];
}

// pad w_m1 [256,16] f32 -> [256,32] bf16 (upper K zero)
__global__ void cvt_pad_m1_kernel(const float* __restrict__ src, __bf16* __restrict__ dst) {
    int i = blockIdx.x * blockDim.x + threadIdx.x;   // 256*32
    int co = i >> 5, ci = i & 31;
    dst[i] = (ci < 16) ? (__bf16)src[co * 16 + ci] : (__bf16)0.0f;
}

// normalize s columns: s [B,16,M] f32 -> snorm [N,32] bf16 (K padded with zeros)
__global__ void norm_kernel(const float* __restrict__ s, __bf16* __restrict__ out) {
    int n = blockIdx.x * blockDim.x + threadIdx.x;
    int b = n >> 11, m = n & 2047;
    const float* col = s + (size_t)b * 16 * MPTS + m;
    float v[16]; float ss = 0.0f;
#pragma unroll
    for (int c = 0; c < 16; ++c) { v[c] = col[(size_t)c * MPTS]; ss += v[c] * v[c]; }
    float inv = rsqrtf(ss);
    v8bf o0, o1, zz;
#pragma unroll
    for (int c = 0; c < 8; ++c) {
        o0[c] = (__bf16)(v[c] * inv);
        o1[c] = (__bf16)(v[c + 8] * inv);
        zz[c] = (__bf16)0.0f;
    }
    __bf16* row = out + (size_t)n * 32;
    *reinterpret_cast<v8bf*>(row)      = o0;
    *reinterpret_cast<v8bf*>(row + 8)  = o1;
    *reinterpret_cast<v8bf*>(row + 16) = zz;
    *reinterpret_cast<v8bf*>(row + 24) = zz;
}

__global__ void bn_finalize_kernel(const float* __restrict__ sum, const float* __restrict__ sq,
                                   const float* __restrict__ g, const float* __restrict__ be,
                                   float* __restrict__ scale, float* __restrict__ shift, int C) {
    int c = blockIdx.x * blockDim.x + threadIdx.x;
    if (c < C) {
        const float invN = 1.0f / (float)N_TOTAL;
        float mu  = sum[c] * invN;
        float var = sq[c] * invN - mu * mu;
        float inv = rsqrtf(var + 1e-5f);
        float sc  = g[c] * inv;
        scale[c] = sc;
        shift[c] = be[c] - mu * sc;
    }
}

// ---------------- fused WMMA GEMM layer ----------------
// actIn [N, CIN] bf16 (raw; optionally BN+ReLU applied on load), W [COUT, CIN] bf16.
// MODE 0: +bias, relu              (map_samples1)
// MODE 1: +bias, clip(-1,1), *z    (map_samples2 + z-modulation)
// MODE 2: +bias, write raw, accumulate per-channel sum/sumsq (conv layers; BN applied by consumer)
template <int CIN, int COUT, int MODE, bool IN_BN>
__global__ __launch_bounds__(256) void gemm_kernel(
    const __bf16* __restrict__ actIn,
    const float* __restrict__ inScale, const float* __restrict__ inShift,
    const __bf16* __restrict__ W, const float* __restrict__ bias,
    const float* __restrict__ z,
    __bf16* __restrict__ actOut,
    float* __restrict__ statSum, float* __restrict__ statSq)
{
    __shared__ float lsum[64];
    __shared__ float lsq[64];
    __shared__ float ldsBN[IN_BN ? 2 * CIN : 2];   // [scale | shift]

    const int tid  = threadIdx.x;
    const int lane = tid & 31;
    const int wave = tid >> 5;
    const int l15  = lane & 15;
    const int half = lane >> 4;
    const int mt   = wave & 3;   // 16-row Cout tile within the 64-row block tile
    const int ng   = wave >> 2;  // 64-col N group within the 128-col block tile
    const int mbase = blockIdx.x * 64 + mt * 16;
    const int nbase = blockIdx.y * 128 + ng * 64;

    if constexpr (IN_BN) {
        // async global->LDS copy of the per-channel affine (ASYNCcnt-tracked, no VGPR staging)
        for (int i = tid; i < CIN; i += 256) {
            unsigned o0 = LDS_OFF(&ldsBN[i]);
            unsigned o1 = LDS_OFF(&ldsBN[CIN + i]);
            asm volatile("global_load_async_to_lds_b32 %0, %1, off"
                         :: "v"(o0), "v"(inScale + i) : "memory");
            asm volatile("global_load_async_to_lds_b32 %0, %1, off"
                         :: "v"(o1), "v"(inShift + i) : "memory");
        }
        asm volatile("s_wait_asynccnt 0x0" ::: "memory");
    }
    if constexpr (MODE == 2) {
        if (tid < 64) { lsum[tid] = 0.0f; lsq[tid] = 0.0f; }
    }
    if (IN_BN || MODE == 2) __syncthreads();

    v8f acc[4];
#pragma unroll
    for (int t = 0; t < 4; ++t)
#pragma unroll
        for (int r = 0; r < 8; ++r) acc[t][r] = 0.0f;

    // hoisted base pointers; kb becomes an immediate offset after full unroll
    const __bf16* aptr = W + (size_t)(mbase + l15) * CIN + half * 16;
    const __bf16* bptr[4];
#pragma unroll
    for (int t = 0; t < 4; ++t)
        bptr[t] = actIn + (size_t)(nbase + t * 16 + l15) * CIN + 16 * half;
    const float* scp = IN_BN ? (ldsBN + 16 * half) : nullptr;
    const float* shp = IN_BN ? (ldsBN + CIN + 16 * half) : nullptr;

    auto loadk = [&](int kb, v8bf& Alo, v8bf& Ahi, v8bf (&Blo)[4], v8bf (&Bhi)[4]) {
        Alo = *reinterpret_cast<const v8bf*>(aptr + kb);
        Ahi = *reinterpret_cast<const v8bf*>(aptr + kb + 8);
#pragma unroll
        for (int t = 0; t < 4; ++t) {
            Blo[t] = *reinterpret_cast<const v8bf*>(bptr[t] + kb);
            Bhi[t] = *reinterpret_cast<const v8bf*>(bptr[t] + kb + 8);
        }
    };

    auto compute = [&](int kb, v8bf Alo, v8bf Ahi, v8bf (&Blo)[4], v8bf (&Bhi)[4]) {
        v16bf a;
#pragma unroll
        for (int j = 0; j < 8; ++j) { a[j] = Alo[j]; a[j + 8] = Ahi[j]; }
        float sc[16], sh[16];
        if constexpr (IN_BN) {
            const float4* sv = reinterpret_cast<const float4*>(scp + kb);
            const float4* hv = reinterpret_cast<const float4*>(shp + kb);
#pragma unroll
            for (int q = 0; q < 4; ++q) {
                *reinterpret_cast<float4*>(&sc[q * 4]) = sv[q];
                *reinterpret_cast<float4*>(&sh[q * 4]) = hv[q];
            }
        }
#pragma unroll
        for (int t = 0; t < 4; ++t) {
            v16bf bfr;
            if constexpr (IN_BN) {
#pragma unroll
                for (int j = 0; j < 8; ++j) {
                    float v0 = fmaf((float)Blo[t][j], sc[j], sh[j]);         v0 = v0 > 0.0f ? v0 : 0.0f;
                    float v1 = fmaf((float)Bhi[t][j], sc[j + 8], sh[j + 8]); v1 = v1 > 0.0f ? v1 : 0.0f;
                    bfr[j] = (__bf16)v0; bfr[j + 8] = (__bf16)v1;
                }
            } else {
#pragma unroll
                for (int j = 0; j < 8; ++j) { bfr[j] = Blo[t][j]; bfr[j + 8] = Bhi[t][j]; }
            }
            acc[t] = __builtin_amdgcn_wmma_f32_16x16x32_bf16(
                false, a, false, bfr, (short)0, acc[t], false, false);
        }
    };

    // 1-deep software pipeline over K: loads for step k+1 issued before WMMAs of step k
    v8bf alo, ahi, blo[4], bhi[4];
    v8bf nalo, nahi, nblo[4], nbhi[4];
    loadk(0, alo, ahi, blo, bhi);
#pragma unroll
    for (int kb = 0; kb < CIN; kb += 32) {
        if (kb + 32 < CIN) loadk(kb + 32, nalo, nahi, nblo, nbhi);
        compute(kb, alo, ahi, blo, bhi);
        if (kb + 32 < CIN) {
            alo = nalo; ahi = nahi;
#pragma unroll
            for (int t = 0; t < 4; ++t) { blo[t] = nblo[t]; bhi[t] = nbhi[t]; }
        }
    }

    // ---- epilogue ----
    const int c0 = mbase + 8 * half;   // first of 8 consecutive output channels this lane holds
    float bi[8];
#pragma unroll
    for (int r = 0; r < 8; ++r) bi[r] = bias[c0 + r];

    float psum[8], psq[8];
#pragma unroll
    for (int r = 0; r < 8; ++r) { psum[r] = 0.0f; psq[r] = 0.0f; }

#pragma unroll
    for (int t = 0; t < 4; ++t) {
        const int n = nbase + t * 16 + l15;
        v8bf o;
#pragma unroll
        for (int r = 0; r < 8; ++r) {
            float v = acc[t][r] + bi[r];
            if constexpr (MODE == 0) { v = v > 0.0f ? v : 0.0f; }
            if constexpr (MODE == 1) {
                v = v < -1.0f ? -1.0f : (v > 1.0f ? 1.0f : v);
                v *= z[(size_t)(n >> 11) * 512 + (c0 + r)];
            }
            if constexpr (MODE == 2) { psum[r] += v; psq[r] += v * v; }
            o[r] = (__bf16)v;
        }
        *reinterpret_cast<v8bf*>(actOut + (size_t)n * COUT + c0) = o;
    }

    if constexpr (MODE == 2) {
        const int lc = mt * 16 + 8 * half;
#pragma unroll
        for (int r = 0; r < 8; ++r) {
            atomicAdd(&lsum[lc + r], psum[r]);
            atomicAdd(&lsq[lc + r],  psq[r]);
        }
        __syncthreads();
        if (tid < 64) {
            unsafeAtomicAdd(&statSum[blockIdx.x * 64 + tid], lsum[tid]);
            unsafeAtomicAdd(&statSq[blockIdx.x * 64 + tid],  lsq[tid]);
        }
    }
}

// ---------------- final 64 -> 3 conv (applies BN3+ReLU) ----------------
__global__ void outconv_kernel(const __bf16* __restrict__ act,
                               const float* __restrict__ scale, const float* __restrict__ shift,
                               const float* __restrict__ wout, const float* __restrict__ bout,
                               float* __restrict__ pts)
{
    int n = blockIdx.x * blockDim.x + threadIdx.x;
    const __bf16* row = act + (size_t)n * 64;
    float a0 = bout[0], a1 = bout[1], a2 = bout[2];
#pragma unroll 8
    for (int c = 0; c < 64; ++c) {
        float v = fmaf((float)row[c], scale[c], shift[c]);
        v = v > 0.0f ? v : 0.0f;
        a0 = fmaf(v, wout[c], a0);
        a1 = fmaf(v, wout[64 + c], a1);
        a2 = fmaf(v, wout[128 + c], a2);
    }
    float* o = pts + (size_t)n * 3;
    o[0] = a0; o[1] = a1; o[2] = a2;
}

// ---------------- kNN graph filter: out = 2*p - mean(8 nearest) ----------------
__global__ __launch_bounds__(256) void knn_kernel(const float* __restrict__ pts,
                                                  float* __restrict__ out)
{
    __shared__ __align__(16) float stage[MPTS * 3];   // raw [m][3] block, async-filled
    __shared__ float px[MPTS], py[MPTS], pz[MPTS], sq[MPTS];
    const int b = blockIdx.x;
    const float* P = pts + (size_t)b * MPTS * 3;

    // async copy of the whole batch's point block into LDS (16B per transfer)
    for (int i = threadIdx.x * 4; i < MPTS * 3; i += 256 * 4) {
        unsigned o = LDS_OFF(&stage[i]);
        asm volatile("global_load_async_to_lds_b128 %0, %1, off"
                     :: "v"(o), "v"(P + i) : "memory");
    }
    asm volatile("s_wait_asynccnt 0x0" ::: "memory");
    __syncthreads();

    for (int i = threadIdx.x; i < MPTS; i += 256) {
        float x = stage[i * 3 + 0], y = stage[i * 3 + 1], zc = stage[i * 3 + 2];
        px[i] = x; py[i] = y; pz[i] = zc;
        sq[i] = x * x + y * y + zc * zc;
    }
    __syncthreads();

    for (int q = threadIdx.x; q < MPTS; q += 256) {
        const float qx = px[q], qy = py[q], qz = pz[q], qs = sq[q];
        float bd[9]; int bi[9];
#pragma unroll
        for (int k = 0; k < 9; ++k) { bd[k] = 3.4e38f; bi[k] = 0; }
        for (int c = 0; c < MPTS; ++c) {
            float d = qs + sq[c] - 2.0f * (qx * px[c] + qy * py[c] + qz * pz[c]);
            if (d < bd[8]) {
                int j = 8;
#pragma unroll
                for (int k = 0; k < 8; ++k) {
                    if (j > 0 && bd[j - 1] > d) { bd[j] = bd[j - 1]; bi[j] = bi[j - 1]; --j; }
                }
                bd[j] = d; bi[j] = c;
            }
        }
        // entry 0 is self (distance ~0); neighbors are entries 1..8
        float sx = 0.0f, sy = 0.0f, sz = 0.0f;
#pragma unroll
        for (int k = 1; k < 9; ++k) { sx += px[bi[k]]; sy += py[bi[k]]; sz += pz[bi[k]]; }
        float* ob = out + (size_t)b * 3 * MPTS;
        ob[0 * MPTS + q] = 2.0f * qx - sx * 0.125f;
        ob[1 * MPTS + q] = 2.0f * qy - sy * 0.125f;
        ob[2 * MPTS + q] = 2.0f * qz - sz * 0.125f;
    }
}

// ---------------- launcher ----------------
extern "C" void kernel_launch(void* const* d_in, const int* in_sizes, int n_in,
                              void* d_out, int out_size, void* d_ws, size_t ws_size,
                              hipStream_t stream)
{
    const float* z    = (const float*)d_in[0];
    const float* s    = (const float*)d_in[1];
    const float* w_m1 = (const float*)d_in[2];
    const float* b_m1 = (const float*)d_in[3];
    const float* w_m2 = (const float*)d_in[4];
    const float* b_m2 = (const float*)d_in[5];
    const float* w_out= (const float*)d_in[6];
    const float* b_out= (const float*)d_in[7];
    const float* w_c[4], *b_c[4], *g_c[4], *be_c[4];
    for (int j = 0; j < 4; ++j) {
        w_c[j]  = (const float*)d_in[8 + 4 * j + 0];
        b_c[j]  = (const float*)d_in[8 + 4 * j + 1];
        g_c[j]  = (const float*)d_in[8 + 4 * j + 2];
        be_c[j] = (const float*)d_in[8 + 4 * j + 3];
    }

    char* ws = (char*)d_ws;
    __bf16* bufA = (__bf16*)(ws + OFF_BUFA);
    __bf16* bufB = (__bf16*)(ws + OFF_BUFB);
    __bf16* wm1  = (__bf16*)(ws + OFF_WM1);
    __bf16* wm2  = (__bf16*)(ws + OFF_WM2);
    __bf16* wc0  = (__bf16*)(ws + OFF_WC0);
    __bf16* wc1  = (__bf16*)(ws + OFF_WC1);
    __bf16* wc2  = (__bf16*)(ws + OFF_WC2);
    __bf16* wc3  = (__bf16*)(ws + OFF_WC3);
    float*  stat = (float*)(ws + OFF_STAT);   // [4][2][512]
    float*  bn   = (float*)(ws + OFF_BN);     // [4][2][512]
    float*  pts  = (float*)(ws + OFF_PTS);
    float*  outp = (float*)d_out;

    auto sumL  = [&](int j) { return stat + j * 1024; };
    auto sqL   = [&](int j) { return stat + j * 1024 + 512; };
    auto scaleL= [&](int j) { return bn + j * 1024; };
    auto shiftL= [&](int j) { return bn + j * 1024 + 512; };

    // 0) zero stats, convert weights to bf16
    zero_stats_kernel<<<16, 256, 0, stream>>>(stat, 4096);
    cvt_pad_m1_kernel<<<32, 256, 0, stream>>>(w_m1, wm1);
    cvt_kernel<<<512,  256, 0, stream>>>(w_m2, wm2, 512 * 256);
    cvt_kernel<<<1024, 256, 0, stream>>>(w_c[0], wc0, 512 * 512);
    cvt_kernel<<<512,  256, 0, stream>>>(w_c[1], wc1, 256 * 512);
    cvt_kernel<<<128,  256, 0, stream>>>(w_c[2], wc2, 128 * 256);
    cvt_kernel<<<32,   256, 0, stream>>>(w_c[3], wc3, 64 * 128);

    // 1) normalize samples -> bufA [N,32]
    norm_kernel<<<N_TOTAL / 256, 256, 0, stream>>>(s, bufA);

    // 2) map_samples1: [N,32]x[256,32] -> bufB, relu
    gemm_kernel<32, 256, 0, false><<<dim3(4, 512), 256, 0, stream>>>(
        bufA, nullptr, nullptr, wm1, b_m1, nullptr, bufB, nullptr, nullptr);

    // 3) map_samples2: hardtanh + z-mod -> bufA
    gemm_kernel<256, 512, 1, false><<<dim3(8, 512), 256, 0, stream>>>(
        bufB, nullptr, nullptr, wm2, b_m2, z, bufA, nullptr, nullptr);

    // 4) conv0 512->512 (raw + stats) -> bufB
    gemm_kernel<512, 512, 2, false><<<dim3(8, 512), 256, 0, stream>>>(
        bufA, nullptr, nullptr, wc0, b_c[0], nullptr, bufB, sumL(0), sqL(0));
    bn_finalize_kernel<<<2, 256, 0, stream>>>(sumL(0), sqL(0), g_c[0], be_c[0], scaleL(0), shiftL(0), 512);

    // 5) conv1 512->256, applies BN0+relu on load -> bufA
    gemm_kernel<512, 256, 2, true><<<dim3(4, 512), 256, 0, stream>>>(
        bufB, scaleL(0), shiftL(0), wc1, b_c[1], nullptr, bufA, sumL(1), sqL(1));
    bn_finalize_kernel<<<1, 256, 0, stream>>>(sumL(1), sqL(1), g_c[1], be_c[1], scaleL(1), shiftL(1), 256);

    // 6) conv2 256->128 -> bufB
    gemm_kernel<256, 128, 2, true><<<dim3(2, 512), 256, 0, stream>>>(
        bufA, scaleL(1), shiftL(1), wc2, b_c[2], nullptr, bufB, sumL(2), sqL(2));
    bn_finalize_kernel<<<1, 128, 0, stream>>>(sumL(2), sqL(2), g_c[2], be_c[2], scaleL(2), shiftL(2), 128);

    // 7) conv3 128->64 -> bufA
    gemm_kernel<128, 64, 2, true><<<dim3(1, 512), 256, 0, stream>>>(
        bufB, scaleL(2), shiftL(2), wc3, b_c[3], nullptr, bufA, sumL(3), sqL(3));
    bn_finalize_kernel<<<1, 64, 0, stream>>>(sumL(3), sqL(3), g_c[3], be_c[3], scaleL(3), shiftL(3), 64);

    // 8) out conv 64->3 (applies BN3+relu) -> pts [N,3]
    outconv_kernel<<<N_TOTAL / 256, 256, 0, stream>>>(bufA, scaleL(3), shiftL(3), w_out, b_out, pts);

    // 9) graph filtering -> d_out [B,3,M]
    knn_kernel<<<NB, 256, 0, stream>>>(pts, outp);
}
